// NCF_62491774157185
// MI455X (gfx1250) — matrix operand order repaired and checked by reading, not compiled, not run on hardware
//
#include <hip/hip_runtime.h>

typedef __attribute__((ext_vector_type(16))) _Float16 v16h;
typedef __attribute__((ext_vector_type(8)))  _Float16 v8h;
typedef __attribute__((ext_vector_type(8)))  float    v8f;

#define HID 64
#define MFD 128
#define NMF 128
#define NM  1024
#define NG  2048

// ---------------------------------------------------------------------------
// Kernel A: fold Wl[:64] into Wc2 -> w2l[64], and scalar c0 = Wl[:64]·bc2 + bl
// ---------------------------------------------------------------------------
__global__ void ncf_fold(const float* __restrict__ Wl, const float* __restrict__ Wc2,
                         const float* __restrict__ bc2, const float* __restrict__ bl,
                         float* __restrict__ w2l, float* __restrict__ c0) {
  int t = threadIdx.x;
  if (t < HID) {
    float s = 0.f;
    for (int k = 0; k < HID; ++k) s += Wl[k] * Wc2[k * HID + t];
    w2l[t] = s;
  }
  if (t == 0) {
    float s = bl[0];
    for (int k = 0; k < HID; ++k) s += Wl[k] * bc2[k];
    c0[0] = s;
  }
}

// ---------------------------------------------------------------------------
// Kernel B: graph MLP.  gu[i,l] = (g @ Wc1[:, :64].T)[i,l] + bc1[l]
//           gh[i,k]    = (f16) g[i,k]
// ---------------------------------------------------------------------------
__global__ __launch_bounds__(256) void ncf_graph(
    const float* __restrict__ X,
    const float* __restrict__ W1, const float* __restrict__ b1,
    const float* __restrict__ W2, const float* __restrict__ b2,
    const float* __restrict__ Wc1, const float* __restrict__ bc1,
    float* __restrict__ gu, _Float16* __restrict__ gh) {
  __shared__ float x[NMF];
  __shared__ float h1[4 * HID];
  __shared__ float row[HID];
  const int i = blockIdx.x, t = threadIdx.x;
  if (t < NMF) x[t] = X[i * NMF + t];
  __syncthreads();
  {
    const float* w = W1 + t * NMF;
    float s = b1[t];
#pragma unroll 4
    for (int k = 0; k < NMF; ++k) s += w[k] * x[k];
    h1[t] = fmaxf(s, 0.f);
  }
  __syncthreads();
  if (t < HID) {
    const float* w = W2 + t * (4 * HID);
    float s = b2[t];
#pragma unroll 4
    for (int k = 0; k < 4 * HID; ++k) s += w[k] * h1[k];
    row[t] = s;
  }
  __syncthreads();
  if (t < HID) {
    const float* w = Wc1 + t * (2 * HID);       // src half of Wc1 row t
    float s = bc1[t];                           // fold bc1 into gu
#pragma unroll 4
    for (int k = 0; k < HID; ++k) s += w[k] * row[k];
    gu[i * HID + t] = s;
    gh[i * HID + t] = (_Float16)row[t];
  }
}

// ---------------------------------------------------------------------------
// Kernel C: model MLP.  mv[j,l]  = (m @ Wc1[:, 64:].T)[j,l]
//           mwh[j,k]   = (f16)( m[j,k] * Wl[64+k] )   (elem-weight folded in)
// ---------------------------------------------------------------------------
__global__ __launch_bounds__(256) void ncf_model(
    const float* __restrict__ X,
    const float* __restrict__ W1, const float* __restrict__ b1,
    const float* __restrict__ W2, const float* __restrict__ b2,
    const float* __restrict__ Wc1, const float* __restrict__ Wl,
    float* __restrict__ mv, _Float16* __restrict__ mwh) {
  __shared__ float x[MFD];
  __shared__ float h1[4 * HID];
  __shared__ float row[HID];
  const int j = blockIdx.x, t = threadIdx.x;
  if (t < MFD) x[t] = X[j * MFD + t];
  __syncthreads();
  {
    const float* w = W1 + t * MFD;
    float s = b1[t];
#pragma unroll 4
    for (int k = 0; k < MFD; ++k) s += w[k] * x[k];
    h1[t] = fmaxf(s, 0.f);
  }
  __syncthreads();
  if (t < HID) {
    const float* w = W2 + t * (4 * HID);
    float s = b2[t];
#pragma unroll 4
    for (int k = 0; k < 4 * HID; ++k) s += w[k] * h1[k];
    row[t] = s;
  }
  __syncthreads();
  if (t < HID) {
    const float* w = Wc1 + t * (2 * HID) + HID; // dst half of Wc1 row t
    float s = 0.f;
#pragma unroll 4
    for (int k = 0; k < HID; ++k) s += w[k] * row[k];
    mv[j * HID + t] = s;
    mwh[j * HID + t] = (_Float16)(row[t] * Wl[HID + t]);
  }
}

// ---------------------------------------------------------------------------
// Main kernel: one wave per 16x16 output tile.
//   score[i,j] = sum_l w2l[l]*relu(gu[i,l]+mv[j,l])   (VALU, LDS-staged)
//              + (gh[i,:]) . (mwh[j,:])               (2x v_wmma_f32_16x16x32_f16)
//              + c0
// Lane->C mapping follows ISA 16x16 f32 C layout: N = lane&15, M = rb + v.
// ---------------------------------------------------------------------------
__global__ __launch_bounds__(32) void ncf_score(
    const float* __restrict__ gu, const float* __restrict__ mv,
    const _Float16* __restrict__ gh, const _Float16* __restrict__ mwh,
    const float* __restrict__ w2l, const float* __restrict__ c0p,
    float* __restrict__ out) {
  __shared__ float sgu[16 * HID];   // broadcast reads -> no padding needed
  __shared__ float smv[16 * 68];    // padded rows (68 floats) -> conflict-free
  __shared__ float sw[HID];
  const int lane = threadIdx.x;
  const int j0 = blockIdx.x * 16;
  const int i0 = blockIdx.y * 16;

  // stage gu tile (rows i0..i0+15 are 1024 contiguous floats)
  {
    const float4* src = (const float4*)(gu + i0 * HID);
    float4* dst = (float4*)sgu;
#pragma unroll
    for (int r = 0; r < 8; ++r) dst[lane + r * 32] = src[lane + r * 32];
  }
  // stage mv tile with padded row stride
  {
    const float4* src = (const float4*)(mv + j0 * HID);
#pragma unroll
    for (int r = 0; r < 8; ++r) {
      int gi = lane + r * 32;                 // float4 index 0..255
      int rowi = gi >> 4, c4 = gi & 15;
      *(float4*)&smv[rowi * 68 + c4 * 4] = src[gi];
    }
  }
  if (lane < 16) ((float4*)sw)[lane] = ((const float4*)w2l)[lane];
  __syncthreads();

  const int n  = lane & 15;
  const int rb = (lane < 16) ? 0 : 8;

  // ---- rank-folded ReLU reduction (VALU) ----
  float acc[8];
#pragma unroll
  for (int v = 0; v < 8; ++v) acc[v] = 0.f;
  const float* mrow = &smv[n * 68];
#pragma unroll 4
  for (int l4 = 0; l4 < 16; ++l4) {
    float4 m4 = *(const float4*)&mrow[l4 * 4];
    float4 w4 = *(const float4*)&sw[l4 * 4];
#pragma unroll
    for (int v = 0; v < 8; ++v) {
      float4 g4 = *(const float4*)&sgu[(rb + v) * HID + l4 * 4];
      acc[v] += w4.x * fmaxf(g4.x + m4.x, 0.f);
      acc[v] += w4.y * fmaxf(g4.y + m4.y, 0.f);
      acc[v] += w4.z * fmaxf(g4.z + m4.z, 0.f);
      acc[v] += w4.w * fmaxf(g4.w + m4.w, 0.f);
    }
  }

  // ---- elem part via WMMA: (g * Wl2) @ m^T, K=64 -> two 16x16x32 f16 ops ----
  // A 16x32 f16 layout: lane<16 holds M=lane, elems 0..15 = K 0..15
  //                     lane>=16            elems 0..15 = K 16..31
  const int alo = (lane < 16) ? 0 : 16;
  const _Float16* ap = gh + (size_t)(i0 + n) * HID + alo;
  // B 32x16 f16 layout: lane<16 holds N=lane, elems 0..15 = K 0..15
  //                     lane>=16 holds N=lane-16, elems 0..15 = K 16..31
  const int bko = (lane < 16) ? 0 : 16;
  const _Float16* bp = mwh + (size_t)(j0 + n) * HID + bko;

  v8h a0lo = *(const v8h*)(ap);
  v8h a0hi = *(const v8h*)(ap + 8);
  v8h a1lo = *(const v8h*)(ap + 32);
  v8h a1hi = *(const v8h*)(ap + 40);
  v8h b0lo = *(const v8h*)(bp);
  v8h b0hi = *(const v8h*)(bp + 8);
  v8h b1lo = *(const v8h*)(bp + 32);
  v8h b1hi = *(const v8h*)(bp + 40);

  v16h A0, A1, B0, B1;
#pragma unroll
  for (int e = 0; e < 8; ++e) {
    A0[e] = a0lo[e]; A0[e + 8] = a0hi[e];
    A1[e] = a1lo[e]; A1[e + 8] = a1hi[e];
    B0[e] = b0lo[e]; B0[e + 8] = b0hi[e];
    B1[e] = b1lo[e]; B1[e + 8] = b1hi[e];
  }

  v8f c;
#pragma unroll
  for (int v = 0; v < 8; ++v) c[v] = acc[v];
  c = __builtin_amdgcn_wmma_f32_16x16x32_f16(false, A0, false, B0, (short)0, c,
                                             false, false);
  c = __builtin_amdgcn_wmma_f32_16x16x32_f16(false, A1, false, B1, (short)0, c,
                                             false, false);

  const float cc = c0p[0];
#pragma unroll
  for (int v = 0; v < 8; ++v)
    out[(size_t)(i0 + rb + v) * NM + (j0 + n)] = c[v] + cc;
}

// ---------------------------------------------------------------------------
extern "C" void kernel_launch(void* const* d_in, const int* in_sizes, int n_in,
                              void* d_out, int out_size, void* d_ws, size_t ws_size,
                              hipStream_t stream) {
  const float* M   = (const float*)d_in[0];
  const float* E   = (const float*)d_in[1];
  const float* Wg1 = (const float*)d_in[2];
  const float* bg1 = (const float*)d_in[3];
  const float* Wg2 = (const float*)d_in[4];
  const float* bg2 = (const float*)d_in[5];
  const float* Wm1 = (const float*)d_in[6];
  const float* bm1 = (const float*)d_in[7];
  const float* Wm2 = (const float*)d_in[8];
  const float* bm2 = (const float*)d_in[9];
  const float* Wc1 = (const float*)d_in[10];
  const float* bc1 = (const float*)d_in[11];
  const float* Wc2 = (const float*)d_in[12];
  const float* bc2 = (const float*)d_in[13];
  const float* Wl  = (const float*)d_in[14];
  const float* bl  = (const float*)d_in[15];
  float* out = (float*)d_out;

  // workspace layout (all offsets 16B-aligned)
  float* gu  = (float*)d_ws;                  // NG*HID f32
  float* mv  = gu + NG * HID;                 // NM*HID f32
  float* w2l = mv + NM * HID;                 // 64 f32
  float* c0  = w2l + HID;                     // 1 f32 (+63 pad)
  _Float16* gh  = (_Float16*)(c0 + 64);       // NG*HID f16
  _Float16* mwh = gh + NG * HID;              // NM*HID f16

  ncf_fold<<<1, 64, 0, stream>>>(Wl, Wc2, bc2, bl, w2l, c0);
  ncf_graph<<<NG, 256, 0, stream>>>(M, Wg1, bg1, Wg2, bg2, Wc1, bc1, gu, gh);
  ncf_model<<<NM, 256, 0, stream>>>(E, Wm1, bm1, Wm2, bm2, Wc1, Wl, mv, mwh);
  ncf_score<<<dim3(NM / 16, NG / 16), 32, 0, stream>>>(gu, mv, gh, mwh, w2l, c0, out);
}
